// CB_Attention_3083786518914
// MI455X (gfx1250) — matrix operand order repaired and checked by reading, not compiled, not run on hardware
//
#include <hip/hip_runtime.h>

// Problem constants (from reference): B=256, H=128, N=2048
#define B_   256
#define H_   128
#define N_   2048
#define K2   256     // 2*H : contraction length of the big GEMM
#define W3H  384     // 3*H : row stride of W
#define NT   64      // n-columns per block in the main kernel
#define CS   264     // LDS column stride in bf16 halfs (256 + pad -> bank spread)

typedef __attribute__((ext_vector_type(16))) __bf16 v16bf;
typedef __attribute__((ext_vector_type(8)))  __bf16 v8bf;
typedef __attribute__((ext_vector_type(8)))  float  v8f;

// Branch-free tanh: 1 - 2/(exp2(2*log2(e)*x) + 1).
// v_mul + v_exp_f32 + v_add + v_rcp_f32 + fma; exact limits at +-inf,
// abs err ~1e-7 (irrelevant after softmax). Avoids ocml's divergent
// EXEC-masked slow path which dominates for |x| >= 0.625 (our common case).
__device__ __forceinline__ float fast_tanh(float x) {
    float e2 = __builtin_amdgcn_exp2f(x * 2.8853900817779268f); // exp(2x)
    return 1.0f - 2.0f * __builtin_amdgcn_rcpf(e2 + 1.0f);
}

// ---------------------------------------------------------------------------
// Kernel 1: pack W[:, 0:2H] (f32) into bf16 WMMA A-fragments.
// Fragment f = m*8+kk covers A rows [m*16, m*16+16), K cols [kk*32, kk*32+32).
// Per-lane half layout for 16x32 bf16 A (ISA 7.12.2):
//   lane t in [0,16):  halfs 0..7 -> K=0..7,   halfs 8..15 -> K=16..23, row = t
//   lane 16+t:         halfs 0..7 -> K=8..15,  halfs 8..15 -> K=24..31, row = t
// Packed so the hot loop does: lane reads 16 contiguous halfs (32B, coalesced).
// ---------------------------------------------------------------------------
__global__ void pack_W_frags(const float* __restrict__ W, __bf16* __restrict__ Afrag) {
    for (int e = threadIdx.x; e < 64 * 512; e += blockDim.x) {
        int f      = e >> 9;        // fragment index (m*8 + kk)
        int within = e & 511;
        int t      = within >> 4;   // lane
        int i      = within & 15;   // half index inside the lane's 32B chunk
        int m  = f >> 3;
        int kk = f & 7;
        int row = m * 16 + (t & 15);
        int klo;
        if (t < 16) klo = (i < 8) ? i          : (16 + (i - 8));
        else        klo = (i < 8) ? (8 + i)    : (24 + (i - 8));
        int k = kk * 32 + klo;      // k < 256 -> first 2H columns of W
        Afrag[e] = (__bf16)W[row * W3H + k];
    }
}

// ---------------------------------------------------------------------------
// Kernel 2: per-(b,h) bias = W[:, 2H:3H] @ decoder_hidden[b,:]  (tiny GEMM)
// ---------------------------------------------------------------------------
__global__ void bias_kernel(const float* __restrict__ W, const float* __restrict__ dec,
                            float* __restrict__ bias) {
    __shared__ float sDec[H_];
    const int b = blockIdx.x;
    const int h = threadIdx.x;
    sDec[h] = dec[b * H_ + h];
    __syncthreads();
    float s = 0.f;
    #pragma unroll 4
    for (int k = 0; k < H_; ++k)
        s += W[h * W3H + 2 * H_ + k] * sDec[k];
    bias[b * H_ + h] = s;
}

// ---------------------------------------------------------------------------
// Kernel 3: main WMMA kernel.
// Block = (b, 64-column n-tile). 256 threads = 8 waves.
//   Stage: X_b tile (256 K-rows x 64 cols, f32) -> bf16 in LDS, column-major
//          (column n contiguous in K, stride CS halfs) so B-fragment loads are
//          K-contiguous per lane:
//            lane t<16  : column j*16+t, K = kk*32 + 0..7, 16..23
//            lane t>=16 : column j*16+(t-16), K = kk*32 + 8..15, 24..31
//   Compute: wave w -> n-subtile j=w&3, M-half mh=w>>2 (4 of 8 M-tiles).
//            8 x v_wmma_f32_16x16x32_bf16 per 16x16 accumulator (K=256).
//            Then fast_tanh(acc + bias) * v, reduce over h (regs + shfl + LDS).
// ---------------------------------------------------------------------------
__global__ __launch_bounds__(256)
void attn_scores_wmma(const float* __restrict__ sh, const float* __restrict__ dh,
                      const __bf16* __restrict__ Afrag, const float* __restrict__ vvec,
                      const float* __restrict__ bias, float* __restrict__ scores) {
    __shared__ __bf16 XL[NT * CS];                 // 33792 B
    __shared__ float  sV[H_], sBias[H_], sPart[128];

    const int b   = blockIdx.x >> 5;               // N/NT = 32 tiles per batch
    const int n0  = (blockIdx.x & 31) * NT;
    const int tid = threadIdx.x;

    if (tid < H_) sV[tid]         = vvec[tid];
    else          sBias[tid - H_] = bias[b * H_ + (tid - H_)];

    // --- Stage X tile (coalesced float4 global loads, bf16 convert, LDS) ---
    const int kr   = tid >> 4;                     // 16 K-rows per iteration
    const int col4 = (tid & 15) * 4;               // 16 threads cover 64 cols
    for (int it = 0; it < 16; ++it) {
        int k = it * 16 + kr;                      // 0..255
        const float* src = (k < H_)
            ? (sh + (size_t)b * H_ * N_ + (size_t)k        * N_)
            : (dh + (size_t)b * H_ * N_ + (size_t)(k - H_) * N_);
        float4 vd = *(const float4*)(src + n0 + col4);
        XL[(col4 + 0) * CS + k] = (__bf16)vd.x;
        XL[(col4 + 1) * CS + k] = (__bf16)vd.y;
        XL[(col4 + 2) * CS + k] = (__bf16)vd.z;
        XL[(col4 + 3) * CS + k] = (__bf16)vd.w;
    }
    __syncthreads();

    // --- WMMA compute ---
    const int w    = tid >> 5;
    const int lane = tid & 31;
    const int j    = w & 3;                        // n-subtile (16 cols)
    const int mh   = w >> 2;                       // which half of M
    const int lhi  = lane >> 4;                    // 0: lanes 0-15, 1: 16-31

    float p = 0.f;                                 // partial score for col (lane&15)
    for (int m = mh * 4; m < mh * 4 + 4; ++m) {
        v8f acc = {0.f, 0.f, 0.f, 0.f, 0.f, 0.f, 0.f, 0.f};
        #pragma unroll
        for (int kk = 0; kk < 8; ++kk) {
            // A fragment: 32B contiguous per lane, coalesced, L2-resident
            v16bf a = *(const v16bf*)(Afrag + ((m * 8 + kk) << 9) + (lane << 4));
            // B fragment from LDS: 16 K-contiguous halfs for this lane's column
            const __bf16* bptr = &XL[(j * 16 + (lane & 15)) * CS + kk * 32 + lhi * 8];
            v8bf blo = *(const v8bf*)(bptr);
            v8bf bhi = *(const v8bf*)(bptr + 16);
            v16bf bm = __builtin_shufflevector(blo, bhi,
                0,1,2,3,4,5,6,7,8,9,10,11,12,13,14,15);
            acc = __builtin_amdgcn_wmma_f32_16x16x32_bf16(
                false, a, false, bm, (short)0, acc, false, false);
        }
        // C/D layout: VGPR r, lanes 0-15 -> M = m*16+r, lanes 16-31 -> M = m*16+8+r
        const int Mb = m * 16 + lhi * 8;
        #pragma unroll
        for (int r = 0; r < 8; ++r) {
            float x = acc[r] + sBias[Mb + r];
            p = fmaf(sV[Mb + r], fast_tanh(x), p);
        }
    }
    // lanes t and t+16 hold the same column: fold halves, then fold M-halves
    p += __shfl_down(p, 16, 32);
    if (lane < 16) sPart[w * 16 + lane] = p;
    __syncthreads();
    if (tid < 64) {
        int jj = tid >> 4, c = tid & 15;
        scores[(size_t)b * N_ + n0 + jj * 16 + c] =
            sPart[jj * 16 + c] + sPart[(jj + 4) * 16 + c];
    }
}

// ---------------------------------------------------------------------------
// Kernel 4: per-row softmax over N=2048 (one block per batch row)
// ---------------------------------------------------------------------------
__global__ __launch_bounds__(256)
void softmax_rows(const float* __restrict__ scores, float* __restrict__ out) {
    __shared__ float red[256];
    const int b = blockIdx.x, tid = threadIdx.x;
    float vals[8];
    #pragma unroll
    for (int i = 0; i < 8; ++i) vals[i] = scores[(size_t)b * N_ + i * 256 + tid];
    float m = vals[0];
    #pragma unroll
    for (int i = 1; i < 8; ++i) m = fmaxf(m, vals[i]);
    red[tid] = m;
    __syncthreads();
    for (int s = 128; s > 0; s >>= 1) {
        if (tid < s) red[tid] = fmaxf(red[tid], red[tid + s]);
        __syncthreads();
    }
    const float M = red[0];
    __syncthreads();
    float e[8], sum = 0.f;
    #pragma unroll
    for (int i = 0; i < 8; ++i) {
        e[i] = __builtin_amdgcn_exp2f((vals[i] - M) * 1.4426950408889634f);
        sum += e[i];
    }
    red[tid] = sum;
    __syncthreads();
    for (int s = 128; s > 0; s >>= 1) {
        if (tid < s) red[tid] += red[tid + s];
        __syncthreads();
    }
    const float inv = 1.0f / red[0];
    #pragma unroll
    for (int i = 0; i < 8; ++i) out[(size_t)b * N_ + i * 256 + tid] = e[i] * inv;
}

// ---------------------------------------------------------------------------
extern "C" void kernel_launch(void* const* d_in, const int* in_sizes, int n_in,
                              void* d_out, int out_size, void* d_ws, size_t ws_size,
                              hipStream_t stream) {
    const float* sh  = (const float*)d_in[0];   // static_hidden  [B,H,N]
    const float* dh  = (const float*)d_in[1];   // dynamic_hidden [B,H,N]
    const float* dec = (const float*)d_in[2];   // decoder_hidden [B,H]
    const float* v   = (const float*)d_in[3];   // v [1,1,H]
    const float* W   = (const float*)d_in[4];   // W [1,H,3H]
    float* out = (float*)d_out;                 // [B,1,N] f32

    char* ws = (char*)d_ws;
    __bf16* Afrag = (__bf16*)ws;                            //  64 KB
    float*  bias  = (float*)(ws + 65536);                   // 128 KB
    float*  scr   = (float*)(ws + 65536 + 131072);          //   2 MB

    pack_W_frags<<<1, 256, 0, stream>>>(W, Afrag);
    bias_kernel<<<B_, H_, 0, stream>>>(W, dec, bias);
    attn_scores_wmma<<<B_ * (N_ / NT), 256, 0, stream>>>(sh, dh, Afrag, v, bias, scr);
    softmax_rows<<<B_, 256, 0, stream>>>(scr, out);
}
